// TransLayer_57707180589413
// MI455X (gfx1250) — matrix operand order repaired and checked by reading, not compiled, not run on hardware
//
#include <hip/hip_runtime.h>
#include <hip/hip_bf16.h>

typedef __attribute__((ext_vector_type(16))) _Float16 v16h;
typedef __attribute__((ext_vector_type(8)))  _Float16 v8h;
typedef __attribute__((ext_vector_type(8)))  float    v8f;
typedef __attribute__((ext_vector_type(2)))  float    v2f;

#define BB      4
#define NSEQ    8192
#define DIMC    512
#define NHEAD   8
#define DH      64
#define MLM     256
#define NBH     32       // BB*NHEAD
#define ROWS    32768    // BB*NSEQ

// ---------------------------------------------------------------- helpers
__device__ __forceinline__ v8f zero8() {
  v8f z;
#pragma unroll
  for (int i = 0; i < 8; ++i) z[i] = 0.f;
  return z;
}

__device__ __forceinline__ v8f wmma16(v16h a, v16h b, v8f c) {
  return __builtin_amdgcn_wmma_f32_16x16x32_f16(false, a, false, b, (short)0, c,
                                                false, false);
}

// A fragment for v_wmma_f32_16x16x32_f16.
// A0 points at row 0 of the 16-row tile; row-major, K contiguous.
// element e of lane L: A[L%16][k0 + (e/8)*16 + (L/16)*8 + e%8]
__device__ __forceinline__ v16h load_a16(const _Float16* A0, int lda, int k0, int lane) {
  const _Float16* p = A0 + (size_t)(lane & 15) * lda + k0 + ((lane >> 4) << 3);
  v8h lo = *(const v8h*)p;
  v8h hi = *(const v8h*)(p + 16);
  v16h r;
#pragma unroll
  for (int i = 0; i < 8; ++i) { r[i] = lo[i]; r[i + 8] = hi[i]; }
  return r;
}

// B fragment (B stored transposed: Bt is N x K row-major, K contiguous).
// element e of lane L: Bt[L%16][k0 + (e/8)*16 + (L/16)*8 + e%8]
__device__ __forceinline__ v16h load_b16(const _Float16* Bt0, int ldb, int k0, int lane) {
  const _Float16* p = Bt0 + (size_t)(lane & 15) * ldb + k0 + ((lane >> 4) << 3);
  v8h lo = *(const v8h*)p;
  v8h hi = *(const v8h*)(p + 16);
  v16h r;
#pragma unroll
  for (int i = 0; i < 8; ++i) { r[i] = lo[i]; r[i + 8] = hi[i]; }
  return r;
}

__device__ __forceinline__ float redmax16(float v) {
#pragma unroll
  for (int m = 1; m < 16; m <<= 1) v = fmaxf(v, __shfl_xor(v, m, 32));
  return v;
}
__device__ __forceinline__ float redsum16(float v) {
#pragma unroll
  for (int m = 1; m < 16; m <<= 1) v += __shfl_xor(v, m, 32);
  return v;
}

// ---------------------------------------------------------------- kernels

// Transpose+convert weights to f16; zero the global-max scalars.
__global__ void k_prep(const float* __restrict__ Wqkv, const float* __restrict__ Wout,
                       _Float16* __restrict__ WqkvT, _Float16* __restrict__ WoutT,
                       float* __restrict__ gmax) {
  int i = blockIdx.x * blockDim.x + threadIdx.x;
  if (i < 1536 * 512) {
    int c = i / 512, k = i % 512;
    WqkvT[i] = (_Float16)Wqkv[(size_t)k * 1536 + c];
  }
  int j = i - 1536 * 512;
  if (j >= 0 && j < 512 * 512) {
    int c = j / 512, k = j % 512;
    WoutT[j] = (_Float16)Wout[(size_t)k * 512 + c];
  }
  if (i < 2) gmax[i] = 0.f;
}

// LayerNorm: one row (512) per block of 256 threads -> f16.
__global__ void __launch_bounds__(256) k_ln(const float* __restrict__ x,
                                            const float* __restrict__ g,
                                            const float* __restrict__ be,
                                            _Float16* __restrict__ h) {
  int row = blockIdx.x, tid = threadIdx.x;
  const float* xr = x + (size_t)row * DIMC;
  float v0 = xr[tid], v1 = xr[tid + 256];
  __shared__ float red[256];
  red[tid] = v0 + v1;
  __syncthreads();
#pragma unroll
  for (int s = 128; s > 0; s >>= 1) {
    if (tid < s) red[tid] += red[tid + s];
    __syncthreads();
  }
  float mu = red[0] * (1.f / 512.f);
  __syncthreads();
  float d0 = v0 - mu, d1 = v1 - mu;
  red[tid] = d0 * d0 + d1 * d1;
  __syncthreads();
#pragma unroll
  for (int s = 128; s > 0; s >>= 1) {
    if (tid < s) red[tid] += red[tid + s];
    __syncthreads();
  }
  float rs = rsqrtf(red[0] * (1.f / 512.f) + 1e-5f);
  _Float16* hr = h + (size_t)row * DIMC;
  hr[tid]       = (_Float16)(d0 * rs * g[tid] + be[tid]);
  hr[tid + 256] = (_Float16)(d1 * rs * g[tid + 256] + be[tid + 256]);
}

// QKV GEMM: h(32768x512) @ WqkvT' -> scatter to q/k/v [bh][n][64] f16 (+vT [bh][64][n]).
// Software-pipelined + unroll 2 (even/odd register sets kill the rotation movs).
__global__ void __launch_bounds__(256) k_qkv(const _Float16* __restrict__ h,
                                             const _Float16* __restrict__ WT,
                                             _Float16* __restrict__ q,
                                             _Float16* __restrict__ k,
                                             _Float16* __restrict__ v,
                                             _Float16* __restrict__ vT) {
  int wave = blockIdx.x * (blockDim.x >> 5) + (threadIdx.x >> 5);
  int lane = threadIdx.x & 31;
  int mt = wave / 24, nt = wave % 24;
  int row0 = mt * 16, col0 = nt * 64;
  const _Float16* A = h + (size_t)row0 * DIMC;
  v8f acc[4];
#pragma unroll
  for (int t = 0; t < 4; ++t) acc[t] = zero8();
  v16h a = load_a16(A, DIMC, 0, lane);
  v16h b[4];
#pragma unroll
  for (int t = 0; t < 4; ++t)
    b[t] = load_b16(WT + (size_t)(col0 + t * 16) * 512, 512, 0, lane);
#pragma unroll 2
  for (int k0 = 0; k0 < 512; k0 += 32) {
    int kn = (k0 + 32) & 511;  // last prefetch wraps (harmless reload)
    v16h an = load_a16(A, DIMC, kn, lane);
    v16h bn[4];
#pragma unroll
    for (int t = 0; t < 4; ++t)
      bn[t] = load_b16(WT + (size_t)(col0 + t * 16) * 512, 512, kn, lane);
#pragma unroll
    for (int t = 0; t < 4; ++t) acc[t] = wmma16(a, b[t], acc[t]);
    a = an;
#pragma unroll
    for (int t = 0; t < 4; ++t) b[t] = bn[t];
  }
  int half = lane >> 4, ln = lane & 15;
#pragma unroll
  for (int t = 0; t < 4; ++t) {
    int col = col0 + t * 16 + ln;
    int which = col >> 9;
    int hcol = col & 511;
    int head = hcol >> 6, dd = hcol & 63;
#pragma unroll
    for (int r = 0; r < 8; ++r) {
      int row = row0 + r + half * 8;
      int b2 = row >> 13, n = row & 8191;
      int bh = b2 * NHEAD + head;
      float vf = acc[t][r];
      size_t idx = ((size_t)bh * NSEQ + n) * DH + dd;
      if (which == 0) q[idx] = (_Float16)(vf * 0.125f);
      else if (which == 1) k[idx] = (_Float16)vf;
      else {
        v[idx] = (_Float16)vf;
        vT[((size_t)bh * DH + dd) * NSEQ + n] = (_Float16)vf;
      }
    }
  }
}

// Landmark means: mean over 32 consecutive rows.
__global__ void k_land(const _Float16* __restrict__ q, const _Float16* __restrict__ k,
                       _Float16* __restrict__ ql, _Float16* __restrict__ kl) {
  int i = blockIdx.x * blockDim.x + threadIdx.x;
  if (i >= NBH * MLM * DH) return;
  int dd = i & 63, j = (i >> 6) & 255, bh = i >> 14;
  const _Float16* qp = q + ((size_t)bh * NSEQ + j * 32) * DH + dd;
  const _Float16* kp = k + ((size_t)bh * NSEQ + j * 32) * DH + dd;
  float sq = 0.f, sk = 0.f;
#pragma unroll 4
  for (int t = 0; t < 32; ++t) { sq += (float)qp[t * DH]; sk += (float)kp[t * DH]; }
  ql[i] = (_Float16)(sq * (1.f / 32.f));
  kl[i] = (_Float16)(sk * (1.f / 32.f));
}

// Depthwise conv (kernel 33 along seq) -> o16[b][n][head*64+d] (write, not add).
__global__ void k_conv(const _Float16* __restrict__ v, const float* __restrict__ Wres,
                       _Float16* __restrict__ o16) {
  int i = blockIdx.x * blockDim.x + threadIdx.x;
  if (i >= NBH * NSEQ * DH) return;
  int dd = i & 63, n = (i >> 6) & 8191, bh = i >> 19;
  int head = bh & 7, b = bh >> 3;
  float s = 0.f;
  const _Float16* vp = v + (size_t)bh * NSEQ * DH + dd;
  for (int t = 0; t < 33; ++t) {
    int nn = n + t - 16;
    if (nn >= 0 && nn < NSEQ) s += Wres[head * 33 + t] * (float)vp[(size_t)nn * DH];
  }
  o16[((size_t)b * NSEQ + n) * DIMC + head * DH + dd] = (_Float16)s;
}

// a2 = softmax(q_l @ k_l^T), f32, [bh][256][256]. One wave = 16 rows.
__global__ void __launch_bounds__(256) k_a2(const _Float16* __restrict__ ql,
                                            const _Float16* __restrict__ kl,
                                            float* __restrict__ a2) {
  int wave = blockIdx.x * (blockDim.x >> 5) + (threadIdx.x >> 5);
  int lane = threadIdx.x & 31;
  int bh = wave >> 4, mt = wave & 15;
  int half = lane >> 4, ln = lane & 15;
  const _Float16* A = ql + ((size_t)bh * MLM + mt * 16) * DH;
  const _Float16* Bt = kl + (size_t)bh * MLM * DH;
  v8f s[16];
#pragma unroll
  for (int t = 0; t < 16; ++t) s[t] = zero8();
  v16h a0 = load_a16(A, DH, 0, lane), a1 = load_a16(A, DH, 32, lane);
#pragma unroll
  for (int tc = 0; tc < 16; tc += 4) {
    v16h bl[4], bh2[4];
#pragma unroll
    for (int u = 0; u < 4; ++u) {
      const _Float16* Bp = Bt + (size_t)(tc + u) * 16 * DH;
      bl[u]  = load_b16(Bp, DH, 0, lane);
      bh2[u] = load_b16(Bp, DH, 32, lane);
    }
#pragma unroll
    for (int u = 0; u < 4; ++u) {
      s[tc + u] = wmma16(a0, bl[u], s[tc + u]);
      s[tc + u] = wmma16(a1, bh2[u], s[tc + u]);
    }
  }
  float* out = a2 + (size_t)bh * MLM * MLM;
#pragma unroll
  for (int r = 0; r < 8; ++r) {
    float m = -1e30f;
#pragma unroll
    for (int t = 0; t < 16; ++t) m = fmaxf(m, s[t][r]);
    m = redmax16(m);
    float ls = 0.f;
#pragma unroll
    for (int t = 0; t < 16; ++t) { float e = __expf(s[t][r] - m); s[t][r] = e; ls += e; }
    float inv = 1.f / redsum16(ls);
    int row = mt * 16 + r + half * 8;
#pragma unroll
    for (int t = 0; t < 16; ++t) out[(size_t)row * MLM + t * 16 + ln] = s[t][r] * inv;
  }
}

// Row/col abs-sum maxima of a2 (global over all bh) via float atomicMax.
__global__ void k_a2sums(const float* __restrict__ a2, float* __restrict__ gmax) {
  int i = blockIdx.x * blockDim.x + threadIdx.x;
  if (i >= NBH * MLM) return;
  int j = i & 255, bh = i >> 8;
  const float* m = a2 + (size_t)bh * 65536;
  float cs = 0.f, rs = 0.f;
  for (int r = 0; r < 256; ++r) { cs += m[(size_t)r * 256 + j]; rs += m[(size_t)j * 256 + r]; }
  atomicMax((unsigned int*)&gmax[0], __float_as_uint(cs));
  atomicMax((unsigned int*)&gmax[1], __float_as_uint(rs));
}

// z0 = a2^T / (gmax0*gmax1)
__global__ void k_zinit(const float* __restrict__ a2, const float* __restrict__ gmax,
                        float* __restrict__ z) {
  int i = blockIdx.x * blockDim.x + threadIdx.x;
  if (i >= NBH * 65536) return;
  int c = i & 255, r = (i >> 8) & 255, bh = i >> 16;
  float inv = 1.f / (gmax[0] * gmax[1]);
  z[i] = a2[(size_t)bh * 65536 + (size_t)c * 256 + r] * inv;
}

// Y = cf*I - X (per 256x256 matrix)
__global__ void k_ew(const float* __restrict__ X, float* __restrict__ Y, float cf) {
  int i = blockIdx.x * blockDim.x + threadIdx.x;
  if (i >= NBH * 65536) return;
  int c = i & 255, r = (i >> 8) & 255;
  Y[i] = (r == c ? cf : 0.f) - X[i];
}

// Batched GEMM: C[bh] = alpha * A[bh](256x256) @ B[bh](256 x ncols), f32 WMMA.
// Wave computes 64(M) x 16(N); double-buffered K loop, unroll 2.
__global__ void __launch_bounds__(256) k_gemm_nn(const float* __restrict__ A,
                                                 const float* __restrict__ B,
                                                 float* __restrict__ C, int ncols,
                                                 float alpha) {
  int wave = blockIdx.x * (blockDim.x >> 5) + (threadIdx.x >> 5);
  int lane = threadIdx.x & 31;
  int half = lane >> 4, ln = lane & 15;
  int ntiles = ncols >> 4;
  int wpb = 4 * ntiles;
  int bh = wave / wpb, wi = wave % wpb;
  int mt = wi / ntiles, nt = wi % ntiles;
  const float* Ab = A + (size_t)bh * 65536 + (size_t)(mt * 64) * 256;
  const float* Bb = B + (size_t)bh * 256 * ncols;
  v8f acc[4];
#pragma unroll
  for (int m = 0; m < 4; ++m) acc[m] = zero8();
  v2f a[4], bb;
#pragma unroll
  for (int m = 0; m < 4; ++m)
    a[m] = *(const v2f*)(Ab + (size_t)(m * 16 + ln) * 256 + half * 2);
  {
    const float* bp = Bb + (size_t)(half * 2) * ncols + nt * 16 + ln;
    bb[0] = bp[0];
    bb[1] = bp[ncols];
  }
#pragma unroll 2
  for (int k0 = 0; k0 < 256; k0 += 4) {
    int kn = (k0 + 4) & 255;  // wrap on last (harmless reload)
    v2f an[4], bn;
#pragma unroll
    for (int m = 0; m < 4; ++m)
      an[m] = *(const v2f*)(Ab + (size_t)(m * 16 + ln) * 256 + kn + half * 2);
    const float* bp = Bb + (size_t)(kn + half * 2) * ncols + nt * 16 + ln;
    bn[0] = bp[0];
    bn[1] = bp[ncols];
#pragma unroll
    for (int m = 0; m < 4; ++m)
      acc[m] = __builtin_amdgcn_wmma_f32_16x16x4_f32(false, a[m], false, bb, (short)0,
                                                     acc[m], false, false);
#pragma unroll
    for (int m = 0; m < 4; ++m) a[m] = an[m];
    bb = bn;
  }
  float* Cb = C + (size_t)bh * 256 * ncols + (size_t)(mt * 64) * ncols + nt * 16 + ln;
#pragma unroll
  for (int m = 0; m < 4; ++m)
#pragma unroll
    for (int r = 0; r < 8; ++r)
      Cb[(size_t)(m * 16 + r + half * 8) * ncols] = alpha * acc[m][r];
}

// Flash-style: w[bh] = softmax(q_l @ k^T) @ v  (online softmax over 8192 keys).
__global__ void __launch_bounds__(128) k_a3v(const _Float16* __restrict__ ql,
                                             const _Float16* __restrict__ kk,
                                             const _Float16* __restrict__ vT,
                                             float* __restrict__ w) {
  __shared__ _Float16 lds[4][16 * 64];
  int wv = threadIdx.x >> 5, lane = threadIdx.x & 31;
  int wave = blockIdx.x * 4 + wv;
  int bh = wave >> 4, mt = wave & 15;
  int half = lane >> 4, ln = lane & 15;
  const _Float16* A = ql + ((size_t)bh * MLM + mt * 16) * DH;
  const _Float16* K = kk + (size_t)bh * NSEQ * DH;
  const _Float16* VT = vT + (size_t)bh * DH * NSEQ;
  _Float16* myp = &lds[wv][0];
  v16h a0 = load_a16(A, DH, 0, lane), a1 = load_a16(A, DH, 32, lane);
  v8f o[4];
#pragma unroll
  for (int t = 0; t < 4; ++t) o[t] = zero8();
  float mrun[8], lrun[8];
#pragma unroll
  for (int r = 0; r < 8; ++r) { mrun[r] = -1e30f; lrun[r] = 0.f; }

  for (int j0 = 0; j0 < NSEQ; j0 += 64) {
    // -- scores: group all 8 B fragment loads before the WMMA burst
    v16h bl[4], bh2[4];
#pragma unroll
    for (int t = 0; t < 4; ++t) {
      const _Float16* Bp = K + (size_t)(j0 + t * 16) * DH;
      bl[t]  = load_b16(Bp, DH, 0, lane);
      bh2[t] = load_b16(Bp, DH, 32, lane);
    }
    v8f s[4];
#pragma unroll
    for (int t = 0; t < 4; ++t) {
      s[t] = zero8();
      s[t] = wmma16(a0, bl[t], s[t]);
      s[t] = wmma16(a1, bh2[t], s[t]);
    }
    // -- V fragments issued early so HBM latency overlaps softmax + LDS staging
    v16h cl[4], ch[4];
#pragma unroll
    for (int t = 0; t < 4; ++t) {
      const _Float16* Bp = VT + (size_t)(t * 16) * NSEQ + j0;
      cl[t] = load_b16(Bp, NSEQ, 0, lane);
      ch[t] = load_b16(Bp, NSEQ, 32, lane);
    }
    float scale[8];
#pragma unroll
    for (int r = 0; r < 8; ++r) {
      float m = fmaxf(fmaxf(s[0][r], s[1][r]), fmaxf(s[2][r], s[3][r]));
      m = redmax16(m);
      float m2 = fmaxf(mrun[r], m);
      scale[r] = __expf(mrun[r] - m2);
      mrun[r] = m2;
      float ls = 0.f;
#pragma unroll
      for (int t = 0; t < 4; ++t) { float e = __expf(s[t][r] - m2); s[t][r] = e; ls += e; }
      lrun[r] = lrun[r] * scale[r] + redsum16(ls);
    }
#pragma unroll
    for (int t = 0; t < 4; ++t)
#pragma unroll
      for (int r = 0; r < 8; ++r)
        myp[(r + half * 8) * 64 + t * 16 + ln] = (_Float16)s[t][r];
    asm volatile("" ::: "memory");
    v16h pa0 = load_a16(myp, 64, 0, lane), pa1 = load_a16(myp, 64, 32, lane);
#pragma unroll
    for (int t = 0; t < 4; ++t) {
#pragma unroll
      for (int r = 0; r < 8; ++r) o[t][r] *= scale[r];
      o[t] = wmma16(pa0, cl[t], o[t]);
      o[t] = wmma16(pa1, ch[t], o[t]);
    }
    asm volatile("" ::: "memory");
  }
  float* W = w + ((size_t)bh * MLM + mt * 16) * DH;
#pragma unroll
  for (int t = 0; t < 4; ++t)
#pragma unroll
    for (int r = 0; r < 8; ++r)
      W[(size_t)(r + half * 8) * DH + t * 16 + ln] = o[t][r] / lrun[r];
}

// t2T16[bh][d][j] = (f16) t2[bh][j][d]
__global__ void k_t2t(const float* __restrict__ t2, _Float16* __restrict__ t2T) {
  int i = blockIdx.x * blockDim.x + threadIdx.x;
  if (i >= NBH * DH * MLM) return;
  int j = i & 255, d = (i >> 8) & 63, bh = i >> 14;
  t2T[i] = (_Float16)t2[((size_t)bh * MLM + j) * DH + d];
}

// Fused: o16 += softmax(q @ k_l^T) @ t2  (adds to conv output in-place).
__global__ void __launch_bounds__(128) k_a1out(const _Float16* __restrict__ q,
                                               const _Float16* __restrict__ kl,
                                               const _Float16* __restrict__ t2T,
                                               _Float16* __restrict__ o16) {
  __shared__ _Float16 lds[4][16 * 256];
  int wv = threadIdx.x >> 5, lane = threadIdx.x & 31;
  int wave = blockIdx.x * 4 + wv;
  int bh = wave >> 9, rt = wave & 511;
  int half = lane >> 4, ln = lane & 15;
  const _Float16* A = q + ((size_t)bh * NSEQ + rt * 16) * DH;
  const _Float16* Bt = kl + (size_t)bh * MLM * DH;
  _Float16* myp = &lds[wv][0];
  v8f s[16];
#pragma unroll
  for (int t = 0; t < 16; ++t) s[t] = zero8();
  v16h a0 = load_a16(A, DH, 0, lane), a1 = load_a16(A, DH, 32, lane);
#pragma unroll
  for (int tc = 0; tc < 16; tc += 4) {
    v16h bl[4], bh2[4];
#pragma unroll
    for (int u = 0; u < 4; ++u) {
      const _Float16* Bp = Bt + (size_t)(tc + u) * 16 * DH;
      bl[u]  = load_b16(Bp, DH, 0, lane);
      bh2[u] = load_b16(Bp, DH, 32, lane);
    }
#pragma unroll
    for (int u = 0; u < 4; ++u) {
      s[tc + u] = wmma16(a0, bl[u], s[tc + u]);
      s[tc + u] = wmma16(a1, bh2[u], s[tc + u]);
    }
  }
#pragma unroll
  for (int r = 0; r < 8; ++r) {
    float m = -1e30f;
#pragma unroll
    for (int t = 0; t < 16; ++t) m = fmaxf(m, s[t][r]);
    m = redmax16(m);
    float ls = 0.f;
#pragma unroll
    for (int t = 0; t < 16; ++t) { float e = __expf(s[t][r] - m); s[t][r] = e; ls += e; }
    float inv = 1.f / redsum16(ls);
#pragma unroll
    for (int t = 0; t < 16; ++t)
      myp[(r + half * 8) * 256 + t * 16 + ln] = (_Float16)(s[t][r] * inv);
  }
  asm volatile("" ::: "memory");
  v8f o[4];
#pragma unroll
  for (int t = 0; t < 4; ++t) o[t] = zero8();
  const _Float16* T = t2T + (size_t)bh * DH * MLM;
  for (int k0 = 0; k0 < 256; k0 += 32) {
    v16h bt[4];
#pragma unroll
    for (int t = 0; t < 4; ++t)
      bt[t] = load_b16(T + (size_t)(t * 16) * MLM, MLM, k0, lane);
    v16h pa = load_a16(myp, 256, k0, lane);
#pragma unroll
    for (int t = 0; t < 4; ++t) o[t] = wmma16(pa, bt[t], o[t]);
  }
  _Float16* Op = o16 + ((size_t)(bh >> 3) * NSEQ + rt * 16) * DIMC + (bh & 7) * DH;
#pragma unroll
  for (int t = 0; t < 4; ++t)
#pragma unroll
    for (int r = 0; r < 8; ++r) {
      size_t idx = (size_t)(r + half * 8) * DIMC + t * 16 + ln;
      Op[idx] = (_Float16)((float)Op[idx] + o[t][r]);
    }
}

// Final projection: out = x + o16 @ W_out + b_out (f32 output). Pipelined, unroll 2.
__global__ void __launch_bounds__(256) k_final(const _Float16* __restrict__ o16,
                                               const _Float16* __restrict__ WoutT,
                                               const float* __restrict__ bout,
                                               const float* __restrict__ x,
                                               float* __restrict__ out) {
  int wave = blockIdx.x * (blockDim.x >> 5) + (threadIdx.x >> 5);
  int lane = threadIdx.x & 31;
  int half = lane >> 4, ln = lane & 15;
  int mt = wave >> 3, nt = wave & 7;
  int row0 = mt * 16, col0 = nt * 64;
  const _Float16* A = o16 + (size_t)row0 * DIMC;
  v8f acc[4];
#pragma unroll
  for (int t = 0; t < 4; ++t) acc[t] = zero8();
  v16h a = load_a16(A, DIMC, 0, lane);
  v16h b[4];
#pragma unroll
  for (int t = 0; t < 4; ++t)
    b[t] = load_b16(WoutT + (size_t)(col0 + t * 16) * 512, 512, 0, lane);
#pragma unroll 2
  for (int k0 = 0; k0 < 512; k0 += 32) {
    int kn = (k0 + 32) & 511;
    v16h an = load_a16(A, DIMC, kn, lane);
    v16h bn[4];
#pragma unroll
    for (int t = 0; t < 4; ++t)
      bn[t] = load_b16(WoutT + (size_t)(col0 + t * 16) * 512, 512, kn, lane);
#pragma unroll
    for (int t = 0; t < 4; ++t) acc[t] = wmma16(a, b[t], acc[t]);
    a = an;
#pragma unroll
    for (int t = 0; t < 4; ++t) b[t] = bn[t];
  }
#pragma unroll
  for (int t = 0; t < 4; ++t) {
    int col = col0 + t * 16 + ln;
    float bo = bout[col];
#pragma unroll
    for (int r = 0; r < 8; ++r) {
      int row = row0 + r + half * 8;
      size_t idx = (size_t)row * DIMC + col;
      out[idx] = x[idx] + acc[t][r] + bo;
    }
  }
}

// ---------------------------------------------------------------- launch
extern "C" void kernel_launch(void* const* d_in, const int* in_sizes, int n_in,
                              void* d_out, int out_size, void* d_ws, size_t ws_size,
                              hipStream_t stream) {
  (void)in_sizes; (void)n_in; (void)out_size; (void)ws_size;
  const float* x     = (const float*)d_in[0];
  const float* gam   = (const float*)d_in[1];
  const float* bet   = (const float*)d_in[2];
  const float* Wqkv  = (const float*)d_in[3];
  const float* Wout  = (const float*)d_in[4];
  const float* bout  = (const float*)d_in[5];
  const float* Wres  = (const float*)d_in[6];
  float* out = (float*)d_out;

  char* p = (char*)d_ws;
  _Float16* h16   = (_Float16*)p; p += (size_t)ROWS * DIMC * 2;          // 32 MB
  _Float16* WqkvT = (_Float16*)p; p += (size_t)1536 * 512 * 2;
  _Float16* WoutT = (_Float16*)p; p += (size_t)512 * 512 * 2;
  _Float16* q16   = (_Float16*)p; p += (size_t)NBH * NSEQ * DH * 2;
  _Float16* k16   = (_Float16*)p; p += (size_t)NBH * NSEQ * DH * 2;
  _Float16* v16   = (_Float16*)p; p += (size_t)NBH * NSEQ * DH * 2;
  _Float16* vT16  = (_Float16*)p; p += (size_t)NBH * DH * NSEQ * 2;
  _Float16* ql16  = (_Float16*)p; p += (size_t)NBH * MLM * DH * 2;
  _Float16* kl16  = (_Float16*)p; p += (size_t)NBH * MLM * DH * 2;
  float* a2f = (float*)p; p += (size_t)NBH * MLM * MLM * 4;
  float* Za  = (float*)p; p += (size_t)NBH * MLM * MLM * 4;
  float* Zb  = (float*)p; p += (size_t)NBH * MLM * MLM * 4;
  float* XZ  = (float*)p; p += (size_t)NBH * MLM * MLM * 4;
  float* T1  = (float*)p; p += (size_t)NBH * MLM * MLM * 4;
  float* T2  = (float*)p; p += (size_t)NBH * MLM * MLM * 4;
  float* w32 = (float*)p; p += (size_t)NBH * MLM * DH * 4;
  float* t2f = (float*)p; p += (size_t)NBH * MLM * DH * 4;
  _Float16* t2T = (_Float16*)p; p += (size_t)NBH * DH * MLM * 2;
  _Float16* o16 = (_Float16*)p; p += (size_t)ROWS * DIMC * 2;
  float* gmax = (float*)p; p += 256;

  k_prep<<<4096, 256, 0, stream>>>(Wqkv, Wout, WqkvT, WoutT, gmax);
  k_ln<<<ROWS, 256, 0, stream>>>(x, gam, bet, h16);
  k_qkv<<<6144, 256, 0, stream>>>(h16, WqkvT, q16, k16, v16, vT16);
  k_land<<<2048, 256, 0, stream>>>(q16, k16, ql16, kl16);
  k_conv<<<65536, 256, 0, stream>>>(v16, Wres, o16);
  k_a2<<<64, 256, 0, stream>>>(ql16, kl16, a2f);
  k_a2sums<<<32, 256, 0, stream>>>(a2f, gmax);
  k_zinit<<<8192, 256, 0, stream>>>(a2f, gmax, Za);

  float* Z = Za;
  float* Zn = Zb;
  for (int it = 0; it < 6; ++it) {
    k_gemm_nn<<<256, 256, 0, stream>>>(a2f, Z, XZ, 256, 1.f);
    k_ew<<<8192, 256, 0, stream>>>(XZ, T1, 7.f);
    k_gemm_nn<<<256, 256, 0, stream>>>(XZ, T1, T2, 256, 1.f);
    k_ew<<<8192, 256, 0, stream>>>(T2, T1, 15.f);
    k_gemm_nn<<<256, 256, 0, stream>>>(XZ, T1, T2, 256, 1.f);
    k_ew<<<8192, 256, 0, stream>>>(T2, T1, 13.f);
    k_gemm_nn<<<256, 256, 0, stream>>>(Z, T1, Zn, 256, 0.25f);
    float* tmp = Z; Z = Zn; Zn = tmp;
  }

  k_a3v<<<128, 128, 0, stream>>>(ql16, k16, vT16, w32);
  k_gemm_nn<<<64, 256, 0, stream>>>(Z, w32, t2f, 64, 1.f);
  k_t2t<<<2048, 256, 0, stream>>>(t2f, t2T);
  k_a1out<<<4096, 128, 0, stream>>>(q16, kl16, t2T, o16);
  k_final<<<2048, 256, 0, stream>>>(o16, WoutT, bout, x, out);
}
